// QNetwork_34943853920988
// MI455X (gfx1250) — matrix operand order repaired and checked by reading, not compiled, not run on hardware
//
#include <hip/hip_runtime.h>
#include <stdint.h>

#define HDIM 256
#define PDIM 8

typedef __attribute__((ext_vector_type(16))) __bf16 v16bf;
typedef __attribute__((ext_vector_type(8)))  float  v8f;

union FragBF {
    v16bf v;
    unsigned int w[8];
};

__device__ __forceinline__ unsigned short f2bf(float f) {
    unsigned int u = __float_as_uint(f);
    u += 0x7FFFu + ((u >> 16) & 1u);           // round-to-nearest-even
    return (unsigned short)(u >> 16);
}

__device__ __forceinline__ float bf2f(unsigned short b) {
    return __uint_as_float(((unsigned int)b) << 16);
}

// Deterministic elementwise dropout mask, p=0.5 keep, scale 2x.
__device__ __forceinline__ float dropmask(unsigned int idx, unsigned int seed) {
    unsigned int x = idx * 0x9E3779B9u + seed * 0x85EBCA6Bu;
    x ^= x >> 16; x *= 0x7FEB352Du;
    x ^= x >> 15; x *= 0x846CA68Bu;
    x ^= x >> 16;
    return (x & 1u) ? 2.0f : 0.0f;
}

// CDNA5 async global->LDS copy (ASYNCcnt path). The LDS destination offset is
// the low 32 bits of the generic pointer (flat LDS aperture: ADDR[31:0] is the
// wave-relative LDS address).
__device__ __forceinline__ void async_g2l_b128(unsigned int ldsOff, const void* g) {
    asm volatile("global_load_async_to_lds_b128 %0, %1, off"
                 :: "v"(ldsOff), "v"((unsigned long long)(uintptr_t)g)
                 : "memory");
}
__device__ __forceinline__ void wait_asynccnt0() {
    asm volatile("s_wait_asynccnt 0" ::: "memory");
}

// ---------------------------------------------------------------- utilities
__global__ void k_fill(float* p, int n, float v) {
    int i = blockIdx.x * blockDim.x + threadIdx.x;
    if (i < n) p[i] = v;
}

__global__ void k_edge_deg(const long long* __restrict__ ei, int E, float* deg) {
    int e = blockIdx.x * blockDim.x + threadIdx.x;
    if (e < E) atomicAdd(&deg[(int)ei[(size_t)E + e]], 1.0f);
}

__global__ void k_rsqrt(float* d, int n) {
    int i = blockIdx.x * blockDim.x + threadIdx.x;
    if (i < n) d[i] = rsqrtf(d[i]);
}

// ------------------------------------------------- layer 0 (rank-1 feature)
__global__ void k_edge_s0(const long long* __restrict__ ei, int E,
                          const float* __restrict__ dinv,
                          const float* __restrict__ x, float* s0) {
    int e = blockIdx.x * blockDim.x + threadIdx.x;
    if (e < E) {
        int s = (int)ei[e];
        int d = (int)ei[(size_t)E + e];
        atomicAdd(&s0[d], dinv[s] * x[s]);
    }
}

__global__ void k_layer0(const float* __restrict__ s0, const float* __restrict__ x,
                         const float* __restrict__ dinv,
                         const float* __restrict__ W0, const float* __restrict__ b0,
                         unsigned short* __restrict__ h /*bf16*/, int N,
                         unsigned int seed) {
    int i = blockIdx.x * blockDim.x + threadIdx.x;
    if (i >= N * HDIM) return;
    int n = i >> 8, f = i & 255;
    float a = dinv[n] * (s0[n] + dinv[n] * x[n]);  // self-loop folded in
    float v = fmaf(a, W0[f], b0[f]);
    v = v > 0.0f ? v : 0.0f;
    h[i] = f2bf(v * dropmask((unsigned)i, seed));
}

// ---------------------------------------------- pack W -> bf16 fragment order
// B-operand lane layout for v_wmma_f32_16x16x32_bf16 (32x16 B):
//   VGPR j<4: K = 2j,2j+1 (+8 for lanes 16-31); VGPR j>=4: K = 16+2(j-4),+1 (+8 for lanes 16-31)
// Packed as Wp[layer][kt][nt][lane][j] : one u32 = two consecutive-K bf16.
__global__ void k_packW(const float* __restrict__ W1, const float* __restrict__ W2,
                        unsigned int* __restrict__ Wp) {
    int i = blockIdx.x * blockDim.x + threadIdx.x;
    if (i >= 2 * 8 * 16 * 32 * 8) return;
    int layer = i >> 15;
    int r = i & 32767;
    int j    =  r        & 7;
    int lane = (r >> 3)  & 31;
    int nt   = (r >> 8)  & 15;
    int kt   = (r >> 12) & 7;
    const float* W = layer ? W2 : W1;
    int k = kt * 32 + (lane >> 4) * 8 + (j < 4 ? 2 * j : 8 + 2 * j);
    int n = nt * 16 + (lane & 15);
    unsigned int lo = f2bf(W[k * HDIM + n]);
    unsigned int hi = f2bf(W[(k + 1) * HDIM + n]);
    Wp[i] = lo | (hi << 16);
}

// ------------------------------------------------------- WMMA GEMM  t = h @ W
// Block: 256 threads = 8 waves. Block owns a 16-row M tile; the bf16 activation
// rows (512B each) are staged straight into LDS with async global->LDS copies
// (no VGPR bounce), then each wave computes a 16x16 tile at
// n0 = (blockIdx.y*8 + wave)*16 via v_wmma_f32_16x16x32_bf16.
__global__ void __launch_bounds__(256) k_gemm(const unsigned short* __restrict__ h /*bf16*/,
                                              const unsigned int* __restrict__ Wp,
                                              float* __restrict__ out, int N) {
    __shared__ unsigned short ldsA[16][264];   // 16 x 256 bf16, padded row (528B)
    int m0   = blockIdx.x * 16;
    int wave = threadIdx.x >> 5;
    int lane = threadIdx.x & 31;

    // async-stage A tile: 512 x 16B chunks, 2 per thread (uniform; EXEC full)
    for (int c = threadIdx.x; c < 512; c += 256) {
        int row = c >> 5;            // 0..15
        int seg = c & 31;            // 16B segment within row
        int gr  = m0 + row; gr = gr < N ? gr : N - 1;
        const void* gsrc = h + (size_t)gr * HDIM + seg * 8;
        unsigned int ldsOff = (unsigned int)(uintptr_t)&ldsA[row][seg * 8];
        async_g2l_b128(ldsOff, gsrc);
    }
    wait_asynccnt0();
    __syncthreads();

    int nt   = blockIdx.y * 8 + wave;
    int arow = lane & 15;
    int hi   = lane >> 4;
    v8f acc = {};
    for (int k0 = 0; k0 < HDIM; k0 += 32) {
        FragBF A, B;
        // A lane layout (16-bit A 16x32): VGPR j<4 -> K=2j,2j+1 (+8 for hi half)
        //                                 VGPR j>=4 -> K=16+2(j-4) (+8 for hi)
#pragma unroll
        for (int j = 0; j < 8; ++j) {
            int kb = (j < 4 ? 2 * j : 16 + 2 * (j - 4)) + hi * 8;
            A.w[j] = *(const unsigned int*)&ldsA[arow][k0 + kb];
        }
        const unsigned int* bp = Wp + ((((size_t)(k0 >> 5) * 16 + nt) * 32 + lane) * 8);
#pragma unroll
        for (int j = 0; j < 8; ++j) B.w[j] = bp[j];
        acc = __builtin_amdgcn_wmma_f32_16x16x32_bf16(
            /*neg_a=*/false, A.v, /*neg_b=*/false, B.v,
            /*c_mod=*/(short)0, acc, /*reuse_a=*/false, /*reuse_b=*/false);
    }
    // C/D layout: VGPR r -> (M = hi*8 + r, N = lane&15)
    int col = nt * 16 + (lane & 15);
#pragma unroll
    for (int r = 0; r < 8; ++r) {
        int row = m0 + hi * 8 + r;
        if (row < N) out[(size_t)row * HDIM + col] = acc[r];
    }
}

// ------------------------------------------------- edge scatter-add (64 ln/edge)
__global__ void k_edge_agg(const long long* __restrict__ ei, int E,
                           const float* __restrict__ dinv,
                           const float* __restrict__ t, float* __restrict__ agg) {
    long long tid = (long long)blockIdx.x * blockDim.x + threadIdx.x;
    if (tid >= (long long)E * 64) return;
    int e = (int)(tid >> 6);
    int l = (int)(tid & 63);
    int s = (int)ei[e];
    int d = (int)ei[(size_t)E + e];
    float norm = dinv[s] * dinv[d];
    const float4 v = *(const float4*)&t[(size_t)s * HDIM + l * 4];
    float* o = &agg[(size_t)d * HDIM + l * 4];
    atomicAdd(o + 0, norm * v.x);
    atomicAdd(o + 1, norm * v.y);
    atomicAdd(o + 2, norm * v.z);
    atomicAdd(o + 3, norm * v.w);
}

// -------------------------------------- bias + self-loop + relu + dropout
__global__ void k_finish(const float* __restrict__ agg, const float* __restrict__ t,
                         const float* __restrict__ dinv, const float* __restrict__ b,
                         unsigned short* __restrict__ h /*bf16*/, int N,
                         unsigned int seed) {
    int i = blockIdx.x * blockDim.x + threadIdx.x;
    if (i >= N * HDIM) return;
    int n = i >> 8, f = i & 255;
    float di = dinv[n];
    float v = agg[i] + di * di * t[i] + b[f];
    v = v > 0.0f ? v : 0.0f;
    h[i] = f2bf(v * dropmask((unsigned)i, seed));
}

// ------------------------------ per-node heads: q[n] = h[n] @ Wq[n] + bq[n]
// One wave per node; lane = kk*8 + p, so each step's 32 lanes read 32
// consecutive floats of Wq (fully coalesced 128B bursts).
__global__ void __launch_bounds__(256) k_head(const unsigned short* __restrict__ h /*bf16*/,
                                              const float* __restrict__ Wq,
                                              const float* __restrict__ bq,
                                              float* __restrict__ q, int N) {
    int wave = threadIdx.x >> 5;
    int lane = threadIdx.x & 31;
    int n = blockIdx.x * 8 + wave;
    if (n >= N) return;
    int kk = lane >> 3;                        // 0..3
    const float* wb = Wq + (size_t)n * (HDIM * PDIM);
    const unsigned short* hb = h + (size_t)n * HDIM;
    float sum = 0.0f;
#pragma unroll 4
    for (int j = 0; j < 64; ++j) {
        if ((j & 15) == 0) __builtin_prefetch(wb + j * 32 + 512, 0, 0);
        int k = j * 4 + kk;
        sum = fmaf(bf2f(hb[k]), wb[j * 32 + lane], sum);  // offset = 32j + 8kk + p
    }
    sum += __shfl_xor(sum, 8, 32);
    sum += __shfl_xor(sum, 16, 32);
    if (lane < PDIM)
        q[(size_t)n * PDIM + lane] = sum + bq[(size_t)n * PDIM + lane];
}

// ---------------------------------------------------------------------------
extern "C" void kernel_launch(void* const* d_in, const int* in_sizes, int n_in,
                              void* d_out, int out_size, void* d_ws, size_t ws_size,
                              hipStream_t stream) {
    const float*     x  = (const float*)d_in[0];
    const long long* ei = (const long long*)d_in[1];
    const float*     W0 = (const float*)d_in[2];
    const float*     b0 = (const float*)d_in[3];
    const float*     W1 = (const float*)d_in[4];
    const float*     b1 = (const float*)d_in[5];
    const float*     W2 = (const float*)d_in[6];
    const float*     b2 = (const float*)d_in[7];
    const float*     Wq = (const float*)d_in[8];
    const float*     bq = (const float*)d_in[9];
    float* q = (float*)d_out;

    int N = in_sizes[0];          // x is [N,1]
    int E = in_sizes[1] / 2;      // edge_index is [2,E]
    size_t NH = (size_t)N * HDIM;

    // workspace layout
    float* ws   = (float*)d_ws;
    float* dinv = ws;                                  // N f32 (also deg temp)
    float* s0   = ws + N;                              // N f32
    float* hT   = ws + 2 * (size_t)N;                  // N*H f32 (GEMM out)
    float* agg  = hT + NH;                             // N*H f32
    unsigned short* hA = (unsigned short*)(agg + NH);  // N*H bf16 activations
    unsigned int*   Wp = (unsigned int*)(hA + NH);     // 2*32768 u32 packed W

    const int TB = 256;
    int gE  = (E + TB - 1) / TB;
    int gNH = (int)((NH + TB - 1) / TB);
    int gN  = (N + TB - 1) / TB;

    // degrees (self-loop = init 1) and dinv
    k_fill<<<gN, TB, 0, stream>>>(dinv, N, 1.0f);
    k_edge_deg<<<gE, TB, 0, stream>>>(ei, E, dinv);
    k_rsqrt<<<gN, TB, 0, stream>>>(dinv, N);

    // layer 0: scalar aggregation then rank-1 expand
    k_fill<<<gN, TB, 0, stream>>>(s0, N, 0.0f);
    k_edge_s0<<<gE, TB, 0, stream>>>(ei, E, dinv, x, s0);
    k_layer0<<<gNH, TB, 0, stream>>>(s0, x, dinv, W0, b0, hA, N, 1u);

    // pack W1/W2 into bf16 fragment-native layout
    k_packW<<<(2 * 32768) / TB, TB, 0, stream>>>(W1, W2, Wp);

    int mTiles = (N + 15) / 16;
    long long aggWork = (long long)E * 64;
    int gAgg = (int)((aggWork + TB - 1) / TB);

    for (int L = 0; L < 2; ++L) {
        const float* bL = (L == 0) ? b1 : b2;
        k_gemm<<<dim3(mTiles, 2), TB, 0, stream>>>(hA, Wp + (size_t)L * 32768, hT, N);
        k_fill<<<gNH, TB, 0, stream>>>(agg, (int)NH, 0.0f);
        k_edge_agg<<<gAgg, TB, 0, stream>>>(ei, E, dinv, hT, agg);
        k_finish<<<gNH, TB, 0, stream>>>(agg, hT, dinv, bL, hA, N, 2u + (unsigned)L);
    }

    // per-node q-value heads
    k_head<<<(N + 7) / 8, TB, 0, stream>>>(hA, Wq, bq, q, N);
}
